// Attension_86620900425758
// MI455X (gfx1250) — matrix operand order repaired and checked by reading, not compiled, not run on hardware
//
#include <hip/hip_runtime.h>
#include <hip/hip_bf16.h>

// ---------------------------------------------------------------------------
// Single-head causal attention for MI455X (gfx1250, wave32, WMMA bf16).
// Phase 0: weights fp32 -> bf16 W^T [192][1024] in ws (one-shot).
// Phase 1: QKV projection, 512 threads / 16 waves per block (low VGPR pressure:
//          6 N-tiles per wave). x-chunk prefetched to regs; weight chunks DMA'd
//          to LDS (global_load_async_to_lds_b128, double-buffered); B-fragments
//          fed through a rotating 4-slot register pipeline.
// Phase 2: flash attention. K / V^T tiles double-buffered via async DMA,
//          overlapped with WMMA compute; online softmax in fp32.
// ws layout (bf16): Q[16384*64] | K[16384*64] | V^T[4][64][4096] | W^T[192][1024]
// ---------------------------------------------------------------------------

#define D_MODEL 1024
#define HEAD    64
#define SEQ     4096
#define BATCH   4
#define NROW    (BATCH * SEQ)   // 16384

typedef __bf16 bf16;
typedef bf16  v16bf __attribute__((ext_vector_type(16)));
typedef float v8f   __attribute__((ext_vector_type(8)));

struct alignas(16) U16x16 { uint4 lo; uint4 hi; };

__device__ __forceinline__ unsigned short f2bfu(float f) {
    unsigned u = __builtin_bit_cast(unsigned, f);
    u += 0x7fffu + ((u >> 16) & 1u);          // round-to-nearest-even
    return (unsigned short)(u >> 16);
}
__device__ __forceinline__ unsigned pack2(float a, float b) {
    return (unsigned)f2bfu(a) | ((unsigned)f2bfu(b) << 16);
}
// A-fragment: elements 0..7 at p[0..7], elements 8..15 at p[16..23]
__device__ __forceinline__ v16bf ld_frag_a(const unsigned short* p) {
    U16x16 u;
    u.lo = *(const uint4*)p;
    u.hi = *(const uint4*)(p + 16);
    return __builtin_bit_cast(v16bf, u);
}
// B-fragment: elements 0..7 at p[0..7], elements 8..15 at p[16..23]
__device__ __forceinline__ v16bf ld_frag_b(const unsigned short* p) {
    U16x16 u;
    u.lo = *(const uint4*)p;
    u.hi = *(const uint4*)(p + 16);
    return __builtin_bit_cast(v16bf, u);
}
// Async DMA: global -> LDS, 16 bytes per lane (ASYNCcnt-tracked)
__device__ __forceinline__ void async_ld_b128(unsigned lds_off, const void* gaddr) {
    asm volatile("global_load_async_to_lds_b128 %0, %1, off"
                 :: "v"(lds_off), "v"((unsigned long long)(uintptr_t)gaddr)
                 : "memory");
}

// ============================================================================
// Kernel 0: one-shot weight convert: Wq|Wk|Wv f32 [1024][64] -> bf16 W^T [192][1024]
// ============================================================================
__global__ __launch_bounds__(256) void wcvt_kernel(
    const float* __restrict__ Wq, const float* __restrict__ Wk,
    const float* __restrict__ Wv, unsigned short* __restrict__ wT)
{
    int f   = blockIdx.x * 256 + threadIdx.x;   // 0..49151 float4 slots
    int mat = f >> 14;
    int rem = f & 16383;
    int k   = rem >> 4;
    int c4  = (rem & 15) * 4;
    const float* W = (mat == 0) ? Wq : (mat == 1) ? Wk : Wv;
    float4 w = *(const float4*)(W + (size_t)k * HEAD + c4);
    wT[((size_t)(mat * 64 + c4 + 0)) * D_MODEL + k] = f2bfu(w.x);
    wT[((size_t)(mat * 64 + c4 + 1)) * D_MODEL + k] = f2bfu(w.y);
    wT[((size_t)(mat * 64 + c4 + 2)) * D_MODEL + k] = f2bfu(w.z);
    wT[((size_t)(mat * 64 + c4 + 3)) * D_MODEL + k] = f2bfu(w.w);
}

// ============================================================================
// Kernel 1: fused QKV projection.  grid = NROW/128, 512 threads (16 waves).
// Wave (mstrip, ngrp): 16-row M strip, 6 of the 12 N-tiles -> 48 acc VGPRs.
// ============================================================================
__global__ __launch_bounds__(512) void qkv_kernel(
    const float* __restrict__ x,
    const float* __restrict__ Bq, const float* __restrict__ Bk,
    const float* __restrict__ Bv,
    const unsigned short* __restrict__ wT,
    unsigned short* __restrict__ ws)
{
    constexpr int KC  = 64;        // K-chunk of d_model per iteration
    constexpr int LXS = KC + 8;    // 72
    constexpr int LWS = KC + 8;    // 72

    __shared__ __align__(16) unsigned short lx [128 * LXS];      // x tile
    __shared__ __align__(16) unsigned short lwT[2][192 * LWS];   // W^T, dbl-buffered

    const int tid    = threadIdx.x;
    const int lane   = tid & 31;
    const int wid    = tid >> 5;     // 0..15
    const int mstrip = wid & 7;
    const int ngrp   = wid >> 3;     // 0..1 -> N tiles ngrp*6 .. +5
    const int nlo    = lane & 15;
    const int half   = lane >> 4;
    const int m0     = blockIdx.x * 128;

    v8f zero = {0.f, 0.f, 0.f, 0.f, 0.f, 0.f, 0.f, 0.f};
    v8f acc[6];
#pragma unroll
    for (int t = 0; t < 6; ++t) acc[t] = zero;

    float4 xr[4];
    // ---- prologue: prefetch x chunk 0 to regs, DMA weight chunk 0 to LDS
#pragma unroll
    for (int it = 0; it < 4; ++it) {
        int f = tid + 512 * it; int row = f >> 4; int c4 = (f & 15) * 4;
        xr[it] = *(const float4*)(x + (size_t)(m0 + row) * D_MODEL + c4);
    }
#pragma unroll
    for (int it = 0; it < 3; ++it) {
        int f = tid + 512 * it; int col = f >> 3; int c8 = (f & 7) * 8;
        async_ld_b128((unsigned)(uintptr_t)&lwT[0][col * LWS + c8],
                      wT + (size_t)col * D_MODEL + c8);
    }

    for (int k0 = 0, buf = 0; k0 < D_MODEL; k0 += KC, buf ^= 1) {
        // ---- publish x chunk from regs (f32 -> bf16, uint2 stores)
#pragma unroll
        for (int it = 0; it < 4; ++it) {
            int f = tid + 512 * it; int row = f >> 4; int c4 = (f & 15) * 4;
            uint2 d; d.x = pack2(xr[it].x, xr[it].y); d.y = pack2(xr[it].z, xr[it].w);
            *(uint2*)&lx[row * LXS + c4] = d;
        }
        asm volatile("s_wait_asynccnt 0" ::: "memory");  // this wave's W DMA done
        __syncthreads();                                 // lx + W tile visible

        // ---- kick next chunk: x -> regs (global), W -> other LDS buffer (DMA)
        if (k0 + KC < D_MODEL) {
#pragma unroll
            for (int it = 0; it < 4; ++it) {
                int f = tid + 512 * it; int row = f >> 4; int c4 = (f & 15) * 4;
                xr[it] = *(const float4*)(x + (size_t)(m0 + row) * D_MODEL + k0 + KC + c4);
            }
#pragma unroll
            for (int it = 0; it < 3; ++it) {
                int f = tid + 512 * it; int col = f >> 3; int c8 = (f & 7) * 8;
                async_ld_b128((unsigned)(uintptr_t)&lwT[buf ^ 1][col * LWS + c8],
                              wT + (size_t)col * D_MODEL + k0 + KC + c8);
            }
        }

        // ---- compute: 12 WMMAs, rotating 4-slot B-fragment pipeline (dist 3)
        const unsigned short* lwb = &lwT[buf][0];
        v16bf a0 = ld_frag_a(&lx[(mstrip * 16 + nlo) * LXS + half * 8 + 0]);
        v16bf a1 = ld_frag_a(&lx[(mstrip * 16 + nlo) * LXS + half * 8 + 32]);
        auto baddr = [&](int u) {
            int t  = ngrp * 6 + ((u < 6) ? u : (u - 6));
            int kk = (u < 6) ? 0 : 32;
            return &lwb[(t * 16 + nlo) * LWS + half * 8 + kk];
        };
        v16bf fb0 = ld_frag_b(baddr(0));
        v16bf fb1 = ld_frag_b(baddr(1));
        v16bf fb2 = ld_frag_b(baddr(2));
        v16bf fb3;
#pragma unroll
        for (int u = 0; u < 12; ++u) {
            if (u + 3 < 12) {                       // load BEFORE wmma: slot (u+3)&3 != u&3
                v16bf ld = ld_frag_b(baddr(u + 3));
                switch ((u + 3) & 3) {
                    case 0: fb0 = ld; break;
                    case 1: fb1 = ld; break;
                    case 2: fb2 = ld; break;
                    default: fb3 = ld; break;
                }
            }
            int t = (u < 6) ? u : (u - 6);
            v16bf bu = ((u & 3) == 0) ? fb0 : ((u & 3) == 1) ? fb1
                     : ((u & 3) == 2) ? fb2 : fb3;
            acc[t] = __builtin_amdgcn_wmma_f32_16x16x32_bf16(
                false, (u < 6) ? a0 : a1, false, bu, (short)0, acc[t], false, false);
        }
        __syncthreads();   // tile reads done before next publish / DMA overwrite
    }

    // ---- epilogue: + bias; Q,K row-major; V transposed [B][hd][seq]
    const float* bias[3] = {Bq, Bk, Bv};
    unsigned short* vT = ws + (size_t)2 * NROW * HEAD;
#pragma unroll
    for (int tl = 0; tl < 6; ++tl) {
        int tt  = ngrp * 6 + tl;
        int mat = tt >> 2;
        int col = (tt & 3) * 16 + nlo;
        float bb = bias[mat][col];
#pragma unroll
        for (int i = 0; i < 8; ++i) {
            int r = m0 + mstrip * 16 + i + 8 * half;
            unsigned short v = f2bfu(acc[tl][i] + bb);
            if (mat < 2) {
                ws[(size_t)mat * NROW * HEAD + (size_t)r * HEAD + col] = v;
            } else {
                int bidx = r >> 12;
                int sidx = r & (SEQ - 1);
                vT[((size_t)bidx * HEAD + col) * SEQ + sidx] = v;
            }
        }
    }
}

// ============================================================================
// Kernel 2: flash attention.  grid = (SEQ/128, BATCH), 256 threads (8 waves).
// ============================================================================
__global__ __launch_bounds__(256) void attn_kernel(
    const unsigned short* __restrict__ ws, float* __restrict__ out)
{
    constexpr int LKS = HEAD + 8;   // 72

    __shared__ __align__(16) unsigned short kt [2][64 * LKS];   // K  [key][hd]
    __shared__ __align__(16) unsigned short vtT[2][64 * LKS];   // V^T[hd][key]
    __shared__ __align__(16) unsigned short pb [8 * 16 * LKS];  // P slab / wave

    const int tid  = threadIdx.x;
    const int lane = tid & 31;
    const int wid  = tid >> 5;
    const int nlo  = lane & 15;
    const int half = lane >> 4;
    const int q0   = blockIdx.x * 128;
    const int b    = blockIdx.y;
    const size_t r0 = (size_t)b * SEQ + q0;

    const unsigned short* qw = ws;
    const unsigned short* kw = ws + (size_t)NROW * HEAD;
    const unsigned short* vb = ws + (size_t)2 * NROW * HEAD + (size_t)b * HEAD * SEQ;

    v16bf aq[2];
#pragma unroll
    for (int c = 0; c < 2; ++c)
        aq[c] = ld_frag_a(qw + (r0 + wid * 16 + nlo) * HEAD + half * 8 + c * 32);

    v8f zero = {0.f, 0.f, 0.f, 0.f, 0.f, 0.f, 0.f, 0.f};
    float mi[8], li[8];
    v8f acc[4];
#pragma unroll
    for (int i = 0; i < 8; ++i) { mi[i] = -3.0e38f; li[i] = 0.f; }
#pragma unroll
    for (int t = 0; t < 4; ++t) acc[t] = zero;

    const int   ntiles = (q0 >> 6) + 2;
    const float SCL    = 0.125f * 1.44269504088896f; // 1/sqrt(64) * log2(e)

    const int row_ = tid >> 3;        // staging: 64 rows x 8 uint4, x2 slots
    const int c8_  = (tid & 7) * 8;
    const int row2 = row_ + 32;

    // ---- prologue DMA: tile 0 into buffer 0
    {
        const unsigned short* kg = kw + ((size_t)b * SEQ) * HEAD;
        async_ld_b128((unsigned)(uintptr_t)&kt [0][row_ * LKS + c8_], kg + (size_t)row_ * HEAD + c8_);
        async_ld_b128((unsigned)(uintptr_t)&kt [0][row2 * LKS + c8_], kg + (size_t)row2 * HEAD + c8_);
        async_ld_b128((unsigned)(uintptr_t)&vtT[0][row_ * LKS + c8_], vb + (size_t)row_ * SEQ + c8_);
        async_ld_b128((unsigned)(uintptr_t)&vtT[0][row2 * LKS + c8_], vb + (size_t)row2 * SEQ + c8_);
    }

    for (int j = 0; j < ntiles; ++j) {
        const int cur = j & 1;
        asm volatile("s_wait_asynccnt 0" ::: "memory");  // this wave's tile-j DMA
        __syncthreads();                                 // tile j visible; prev reads done
        // ---- kick DMA for tile j+1 into the other buffer
        if (j + 1 < ntiles) {
            const int nxt = cur ^ 1;
            const unsigned short* kg = kw + ((size_t)b * SEQ + (size_t)(j + 1) * 64) * HEAD;
            const unsigned short* vg = vb + (size_t)(j + 1) * 64;
            async_ld_b128((unsigned)(uintptr_t)&kt [nxt][row_ * LKS + c8_], kg + (size_t)row_ * HEAD + c8_);
            async_ld_b128((unsigned)(uintptr_t)&kt [nxt][row2 * LKS + c8_], kg + (size_t)row2 * HEAD + c8_);
            async_ld_b128((unsigned)(uintptr_t)&vtT[nxt][row_ * LKS + c8_], vg + (size_t)row_ * SEQ + c8_);
            async_ld_b128((unsigned)(uintptr_t)&vtT[nxt][row2 * LKS + c8_], vg + (size_t)row2 * SEQ + c8_);
        }
        const unsigned short* ktc = &kt [cur][0];
        const unsigned short* vtc = &vtT[cur][0];

        // ---- S = Q K^T : 8 WMMAs, rotating 4-slot B pipeline (dist 3)
        v8f s[4];
#pragma unroll
        for (int t = 0; t < 4; ++t) s[t] = zero;
        auto kaddr = [&](int u) {
            int t = u >> 1, ch = u & 1;
            return &ktc[(t * 16 + nlo) * LKS + ch * 32 + half * 8];
        };
        {
            v16bf fb0 = ld_frag_b(kaddr(0));
            v16bf fb1 = ld_frag_b(kaddr(1));
            v16bf fb2 = ld_frag_b(kaddr(2));
            v16bf fb3;
#pragma unroll
            for (int u = 0; u < 8; ++u) {
                if (u + 3 < 8) {
                    v16bf ld = ld_frag_b(kaddr(u + 3));
                    switch ((u + 3) & 3) {
                        case 0: fb0 = ld; break;
                        case 1: fb1 = ld; break;
                        case 2: fb2 = ld; break;
                        default: fb3 = ld; break;
                    }
                }
                int t = u >> 1, ch = u & 1;
                v16bf bu = ((u & 3) == 0) ? fb0 : ((u & 3) == 1) ? fb1
                         : ((u & 3) == 2) ? fb2 : fb3;
                s[t] = __builtin_amdgcn_wmma_f32_16x16x32_bf16(
                    false, aq[ch], false, bu, (short)0, s[t], false, false);
            }
        }

        // ---- causal mask + online softmax
        float pv[4][8];
#pragma unroll
        for (int i = 0; i < 8; ++i) {
            int   qrow = q0 + wid * 16 + i + 8 * half;
            float pm   = -3.0e38f;
#pragma unroll
            for (int t = 0; t < 4; ++t) {
                int   key = j * 64 + t * 16 + nlo;
                float sv  = (key <= qrow) ? s[t][i] * SCL : -3.0e38f;
                pv[t][i]  = sv;
                pm        = fmaxf(pm, sv);
            }
#pragma unroll
            for (int d = 1; d < 16; d <<= 1)
                pm = fmaxf(pm, __shfl_xor(pm, d, 32));
            float mnew  = fmaxf(mi[i], pm);
            float alpha = exp2f(mi[i] - mnew);
            float rs    = 0.f;
#pragma unroll
            for (int t = 0; t < 4; ++t) {
                float p  = exp2f(pv[t][i] - mnew);
                pv[t][i] = p;
                rs      += p;
            }
#pragma unroll
            for (int d = 1; d < 16; d <<= 1)
                rs += __shfl_xor(rs, d, 32);
            li[i] = li[i] * alpha + rs;
            mi[i] = mnew;
#pragma unroll
            for (int t = 0; t < 4; ++t) acc[t][i] *= alpha;
        }

        // ---- P -> LDS (C-layout -> A-layout), O += P V (8 WMMAs, pipelined)
        unsigned short* pw = &pb[wid * 16 * LKS];
#pragma unroll
        for (int t = 0; t < 4; ++t)
#pragma unroll
            for (int i = 0; i < 8; ++i)
                pw[(i + 8 * half) * LKS + t * 16 + nlo] = f2bfu(pv[t][i]);
        asm volatile("s_wait_dscnt 0" ::: "memory");   // own-wave LDS RAW

        v16bf ap0 = ld_frag_a(&pw[nlo * LKS + half * 8 + 0]);
        v16bf ap1 = ld_frag_a(&pw[nlo * LKS + half * 8 + 32]);
        auto vaddr = [&](int u) {
            int ch = u >> 2, tn = u & 3;
            return &vtc[(tn * 16 + nlo) * LKS + ch * 32 + half * 8];
        };
        {
            v16bf fb0 = ld_frag_b(vaddr(0));
            v16bf fb1 = ld_frag_b(vaddr(1));
            v16bf fb2 = ld_frag_b(vaddr(2));
            v16bf fb3;
#pragma unroll
            for (int u = 0; u < 8; ++u) {
                if (u + 3 < 8) {
                    v16bf ld = ld_frag_b(vaddr(u + 3));
                    switch ((u + 3) & 3) {
                        case 0: fb0 = ld; break;
                        case 1: fb1 = ld; break;
                        case 2: fb2 = ld; break;
                        default: fb3 = ld; break;
                    }
                }
                int ch = u >> 2, tn = u & 3;
                v16bf bu = ((u & 3) == 0) ? fb0 : ((u & 3) == 1) ? fb1
                         : ((u & 3) == 2) ? fb2 : fb3;
                acc[tn] = __builtin_amdgcn_wmma_f32_16x16x32_bf16(
                    false, ch ? ap1 : ap0, false, bu, (short)0, acc[tn], false, false);
            }
        }
    }

    // ---- epilogue: O = acc / l  (fp32 out)
#pragma unroll
    for (int t = 0; t < 4; ++t)
#pragma unroll
        for (int i = 0; i < 8; ++i) {
            float inv = 1.0f / li[i];
            size_t r  = r0 + wid * 16 + i + 8 * half;
            out[r * HEAD + t * 16 + nlo] = acc[t][i] * inv;
        }
}

// ============================================================================
extern "C" void kernel_launch(void* const* d_in, const int* in_sizes, int n_in,
                              void* d_out, int out_size, void* d_ws, size_t ws_size,
                              hipStream_t stream) {
    (void)in_sizes; (void)n_in; (void)out_size; (void)ws_size;
    const float* x  = (const float*)d_in[0];
    const float* Wq = (const float*)d_in[1];
    const float* Bq = (const float*)d_in[2];
    const float* Wk = (const float*)d_in[3];
    const float* Bk = (const float*)d_in[4];
    const float* Wv = (const float*)d_in[5];
    const float* Bv = (const float*)d_in[6];
    unsigned short* ws = (unsigned short*)d_ws;
    unsigned short* wT = ws + (size_t)3 * NROW * HEAD;   // bf16 W^T [192][1024]

    wcvt_kernel<<<dim3(192), dim3(256), 0, stream>>>(Wq, Wk, Wv, wT);
    qkv_kernel<<<dim3(NROW / 128), dim3(512), 0, stream>>>(
        x, Bq, Bk, Bv, wT, ws);
    attn_kernel<<<dim3(SEQ / 128, BATCH), dim3(256), 0, stream>>>(
        ws, (float*)d_out);
}
